// SocialLSTM_9620726743235
// MI455X (gfx1250) — matrix operand order repaired and checked by reading, not compiled, not run on hardware
//
#include <hip/hip_runtime.h>
#include <hip/hip_bf16.h>

typedef _Float16 f16;
typedef __attribute__((ext_vector_type(16))) _Float16 v16h;
typedef __attribute__((ext_vector_type(8)))  float    v8f;

#define N_AG   8192
#define MM     32
#define OBS_L  8
#define PRED_L 12
#define WPB    4           // waves per block
#define AGW    16          // agents per wave

// ---- workspace layout (bytes) ----
#define WS_ENCW  0                          // f16 [512][192]  (= [Wih;Whh]^T, enc)
#define WS_DECW  (WS_ENCW + 512*192*2)      // f16 [512][320]  (= [Wih;Whh]^T, dec)
#define WS_SWH   (WS_DECW + 512*320*2)      // f16 [128][128]  sp_soc_W[0:128]^T
#define WS_SWP   (WS_SWH + 128*128*2)       // f16 [128][64]   sp_soc_W[128:192]^T
#define WS_ENCB  (WS_SWP + 128*64*2)        // f32 [512] bih+bhh
#define WS_DECB  (WS_ENCB + 512*4)          // f32 [512]

__device__ inline v8f vzero() {
    v8f z = {0.f,0.f,0.f,0.f,0.f,0.f,0.f,0.f};
    return z;
}
__device__ inline float sigm(float x)   { return 1.0f / (1.0f + __expf(-x)); }
__device__ inline float tanh_f(float x) { return 1.0f - 2.0f / (__expf(2.0f * x) + 1.0f); }

union FragU { v16h v; uint4 q[2]; };

// A fragment (16-bit A 16x32): lane m=L&15, kh=L>>4; regs0-3: k=kh*8+0..7, regs4-7: k=16+kh*8+0..7
__device__ inline v16h load_a_frag(const f16* act, int ldk, int lane) {
    const int m = lane & 15, kh = lane >> 4;
    const f16* p = act + m * ldk + kh * 8;
    FragU f;
    f.q[0] = *(const uint4*)(p);
    f.q[1] = *(const uint4*)(p + 16);
    return f.v;
}
// B fragment (16-bit B 32x16) from transposed weights wT[N][ldk]: lane n=L&15, kh=L>>4; regs0-3: k=kh*8+0..7, regs4-7: k=16+kh*8+0..7
__device__ inline v16h load_b_frag(const f16* wT, int ldk, int lane) {
    const int n = lane & 15, kh = lane >> 4;
    const f16* p = wT + n * ldk + kh * 8;
    FragU f;
    f.q[0] = *(const uint4*)(p);
    f.q[1] = *(const uint4*)(p + 16);
    return f.v;
}
__device__ inline v8f wmma16(v16h a, v16h b, v8f c) {
    return __builtin_amdgcn_wmma_f32_16x16x32_f16(false, a, false, b, (short)0, c, false, false);
}

// ------------------------------------------------------------------ prep: pack weights f16-transposed
__global__ void prep_kernel(const float* encWih, const float* encWhh,
                            const float* decWih, const float* decWhh,
                            const float* socW,
                            const float* ebih, const float* ebhh,
                            const float* dbih, const float* dbhh,
                            char* ws) {
    f16* encW = (f16*)(ws + WS_ENCW);
    f16* decW = (f16*)(ws + WS_DECW);
    f16* sWh  = (f16*)(ws + WS_SWH);
    f16* sWp  = (f16*)(ws + WS_SWP);
    float* eB = (float*)(ws + WS_ENCB);
    float* dB = (float*)(ws + WS_DECB);
    const int tot = 512*192 + 512*320 + 128*128 + 128*64 + 512 + 512;
    for (int i = blockIdx.x * blockDim.x + threadIdx.x; i < tot; i += gridDim.x * blockDim.x) {
        int r = i;
        if (r < 512*192) { int n = r / 192, k = r % 192;
            encW[n*192 + k] = (f16)(k < 64 ? encWih[k*512 + n] : encWhh[(k-64)*512 + n]); continue; }
        r -= 512*192;
        if (r < 512*320) { int n = r / 320, k = r % 320;
            decW[n*320 + k] = (f16)(k < 192 ? decWih[k*512 + n] : decWhh[(k-192)*512 + n]); continue; }
        r -= 512*320;
        if (r < 128*128) { int n = r / 128, k = r % 128;
            sWh[n*128 + k] = (f16)socW[k*128 + n]; continue; }
        r -= 128*128;
        if (r < 128*64) { int n = r / 64, k = r % 64;
            sWp[n*64 + k] = (f16)socW[(128 + k)*128 + n]; continue; }
        r -= 128*64;
        if (r < 512) { eB[r] = ebih[r] + ebhh[r]; continue; }
        r -= 512;
        dB[r] = dbih[r] + dbhh[r];
    }
}

// ------------------------------------------------------------------ fused persistent Social-LSTM
__global__ __launch_bounds__(WPB * 32)
void social_lstm_kernel(const float* __restrict__ obs, const float* __restrict__ nb_obs,
                        const int* __restrict__ nb_mask,
                        const float* __restrict__ weW, const float* __restrict__ web,
                        const float* __restrict__ spW, const float* __restrict__ spb,
                        const float* __restrict__ socb,
                        const float* __restrict__ wdW, const float* __restrict__ wdb,
                        const float* __restrict__ outW, const float* __restrict__ outb,
                        const char* __restrict__ ws, float* __restrict__ out) {
    // ---- block-shared weights ----
    __shared__ f16   shWh[128 * 128];
    __shared__ f16   shWp[128 * 64];
    __shared__ float shEncB[512];
    __shared__ float shDecB[512];
    __shared__ float shSb[128];
    __shared__ float shSpW[128], shSpb[64];
    __shared__ float shWeW[128], shWeb[64];
    __shared__ float shWdW[128], shWdb[64];
    __shared__ float shOutW[640], shOutb[8];
    // ---- per-wave working set ----
    __shared__ f16   shAbuf[WPB][16 * 320];   // enc: [emb(64)|h(128)]  dec: [emb|social|h]
    __shared__ f16   shPf[WPB][32 * 64];      // pos_feat rows; reused as f16 h-staging (htmp)
    __shared__ float shSH[WPB][16 * 128];     // h @ sWh (per agent)
    __shared__ float shSoc[WPB][16 * 128];    // enc: social   dec: h(f32) for head
    __shared__ float shC[WPB][16 * 128];      // LSTM cell state (f32)
    __shared__ float shCur[WPB][16][2];
    __shared__ unsigned shVm[WPB][16];
    __shared__ float shOrel[WPB][16][OBS_L][2];
    __shared__ float shOrig[WPB][16][2];

    const int tid  = threadIdx.x;
    const int lane = tid & 31;
    const int wv   = tid >> 5;
    const int col  = lane & 15;
    const int kh   = lane >> 4;
    const f16* gEncW = (const f16*)(ws + WS_ENCW);
    const f16* gDecW = (const f16*)(ws + WS_DECW);

    for (int i = tid; i < 128*128; i += WPB*32) shWh[i] = ((const f16*)(ws + WS_SWH))[i];
    for (int i = tid; i < 128*64;  i += WPB*32) shWp[i] = ((const f16*)(ws + WS_SWP))[i];
    for (int i = tid; i < 512; i += WPB*32) {
        shEncB[i] = ((const float*)(ws + WS_ENCB))[i];
        shDecB[i] = ((const float*)(ws + WS_DECB))[i];
    }
    for (int i = tid; i < 128; i += WPB*32) {
        shSb[i] = socb[i]; shSpW[i] = spW[i]; shWeW[i] = weW[i]; shWdW[i] = wdW[i];
    }
    for (int i = tid; i < 64; i += WPB*32) {
        shSpb[i] = spb[i]; shWeb[i] = web[i]; shWdb[i] = wdb[i];
    }
    for (int i = tid; i < 640; i += WPB*32) shOutW[i] = outW[i];
    if (tid < 5) shOutb[tid] = outb[tid];
    __syncthreads();

    f16*   abuf = shAbuf[wv];
    f16*   pf   = shPf[wv];
    f16*   htmp = shPf[wv];     // alias: pos_feat no longer needed once gates run
    float* sH   = shSH[wv];
    float* soc  = shSoc[wv];
    float* cbuf = shC[wv];
    const int agBase = blockIdx.x * (WPB * AGW) + wv * AGW;

    // per-wave init: relative obs, origin, neighbor-mask bits, zero h & c
    for (int i = lane; i < 16 * OBS_L; i += 32) {
        int a = i >> 3, t = i & 7, aG = agBase + a;
        float ox = obs[(aG*OBS_L + OBS_L-1)*2 + 0];
        float oy = obs[(aG*OBS_L + OBS_L-1)*2 + 1];
        shOrel[wv][a][t][0] = obs[(aG*OBS_L + t)*2 + 0] - ox;
        shOrel[wv][a][t][1] = obs[(aG*OBS_L + t)*2 + 1] - oy;
        if (t == 0) { shOrig[wv][a][0] = ox; shOrig[wv][a][1] = oy; }
    }
    for (int a = 0; a < 16; ++a) {
        bool mk = nb_mask[(agBase + a) * MM + lane] != 0;
        unsigned mb = (unsigned)__ballot(mk);
        if (lane == 0) shVm[wv][a] = mb;
    }
    for (int i = lane; i < 16 * 128; i += 32) {
        abuf[(i >> 7) * 320 + 64 + (i & 127)] = (f16)0.f;
        cbuf[i] = 0.f;
    }

    // ============================== ENCODER ==============================
    for (int t = 0; t < OBS_L; ++t) {
        // (1) sH = h @ sWh^T   (per-agent, hoisted out of the neighbor loop)
        {
            v16h haf[4];
            #pragma unroll
            for (int kt = 0; kt < 4; ++kt) haf[kt] = load_a_frag(abuf + 64 + kt*32, 320, lane);
            for (int jt = 0; jt < 8; ++jt) {
                v8f acc = vzero();
                #pragma unroll
                for (int kt = 0; kt < 4; ++kt)
                    acc = wmma16(haf[kt], load_b_frag(shWh + (jt*16)*128 + kt*32, 128, lane), acc);
                #pragma unroll
                for (int v = 0; v < 8; ++v) sH[(v + 8*kh)*128 + jt*16 + col] = acc[v];
            }
        }
        // (2) social pooling: per agent, 32 neighbor rows through sWp
        for (int a = 0; a < 16; ++a) {
            const int aG = agBase + a;
            const float fx = shOrel[wv][a][t][0], fy = shOrel[wv][a][t][1];
            const int m = lane;
            const float* nb = nb_obs + (size_t)((aG*MM + m)*OBS_L) * 2;
            float rx = (nb[t*2+0] - nb[(OBS_L-1)*2+0]) - fx;
            float ry = (nb[t*2+1] - nb[(OBS_L-1)*2+1]) - fy;
            bool val = (rx*rx + ry*ry <= 4.0f) && ((shVm[wv][a] >> m) & 1u);
            unsigned vm = (unsigned)__ballot(val);
            for (int e = 0; e < 64; ++e)
                pf[m*64 + e] = (f16)fmaxf(0.f, rx*shSpW[e] + ry*shSpW[64 + e] + shSpb[e]);
            v16h saf[4];
            #pragma unroll
            for (int q = 0; q < 4; ++q)   // q = mt*2 + kt
                saf[q] = load_a_frag(pf + (q >> 1)*16*64 + (q & 1)*32, 64, lane);
            for (int jt = 0; jt < 8; ++jt) {
                float best = 0.f;
                float sh_sb = sH[a*128 + jt*16 + col] + shSb[jt*16 + col];
                #pragma unroll
                for (int mt = 0; mt < 2; ++mt) {
                    v8f acc = vzero();
                    #pragma unroll
                    for (int kt = 0; kt < 2; ++kt)
                        acc = wmma16(saf[mt*2 + kt],
                                     load_b_frag(shWp + (jt*16)*64 + kt*32, 64, lane), acc);
                    #pragma unroll
                    for (int v = 0; v < 8; ++v) {
                        int mrow = mt*16 + v + 8*kh;
                        float p = fmaxf(0.f, acc[v] + sh_sb);
                        p = ((vm >> mrow) & 1u) ? p : 0.f;
                        best = fmaxf(best, p);
                    }
                }
                best = fmaxf(best, __shfl_xor(best, 16));
                if (lane < 16) soc[a*128 + jt*16 + col] = best;  // relu>=0 => max includes zeros
            }
        }
        // (3) input embedding
        for (int i = lane; i < 16 * 64; i += 32) {
            int a = i >> 6, e = i & 63;
            float x = shOrel[wv][a][t][0], y = shOrel[wv][a][t][1];
            abuf[a*320 + e] = (f16)fmaxf(0.f, x*shWeW[e] + y*shWeW[64 + e] + shWeb[e]);
        }
        // (4) fused gates: [emb|h](K=192) @ [Wih;Whh]^T, i/f/g/o per 16-col tile
        {
            v16h gaf[6];
            #pragma unroll
            for (int kt = 0; kt < 6; ++kt) gaf[kt] = load_a_frag(abuf + kt*32, 320, lane);
            for (int j = 0; j < 8; ++j) {
                v8f ai = vzero(), af = vzero(), ag = vzero(), ao = vzero();
                #pragma unroll
                for (int kt = 0; kt < 6; ++kt) {
                    ai = wmma16(gaf[kt], load_b_frag(gEncW + (  0 + j*16)*192 + kt*32, 192, lane), ai);
                    af = wmma16(gaf[kt], load_b_frag(gEncW + (128 + j*16)*192 + kt*32, 192, lane), af);
                    ag = wmma16(gaf[kt], load_b_frag(gEncW + (256 + j*16)*192 + kt*32, 192, lane), ag);
                    ao = wmma16(gaf[kt], load_b_frag(gEncW + (384 + j*16)*192 + kt*32, 192, lane), ao);
                }
                const int u = j*16 + col;
                const float bi = shEncB[u], bf = shEncB[128+u], bg = shEncB[256+u], bo = shEncB[384+u];
                #pragma unroll
                for (int v = 0; v < 8; ++v) {
                    int mrow = v + 8*kh;
                    float cold = cbuf[mrow*128 + u];
                    float cn = sigm(af[v] + bf) * cold + sigm(ai[v] + bi) * tanh_f(ag[v] + bg);
                    cbuf[mrow*128 + u] = cn;
                    htmp[mrow*128 + u] = (f16)(sigm(ao[v] + bo) * tanh_f(cn) + 0.3f * soc[mrow*128 + u]);
                }
            }
        }
        // (5) publish h (f16) after all gate K-reads of old h are done
        for (int i = lane; i < 16 * 128; i += 32)
            abuf[(i >> 7) * 320 + 64 + (i & 127)] = htmp[i];
    }

    // ====================== ENCODER -> DECODER transition ======================
    // sH := h_final @ sWh^T (fixed for all decoder steps, since nb_h_last is fixed)
    {
        v16h haf[4];
        #pragma unroll
        for (int kt = 0; kt < 4; ++kt) haf[kt] = load_a_frag(abuf + 64 + kt*32, 320, lane);
        for (int jt = 0; jt < 8; ++jt) {
            v8f acc = vzero();
            #pragma unroll
            for (int kt = 0; kt < 4; ++kt)
                acc = wmma16(haf[kt], load_b_frag(shWh + (jt*16)*128 + kt*32, 128, lane), acc);
            #pragma unroll
            for (int v = 0; v < 8; ++v) sH[(v + 8*kh)*128 + jt*16 + col] = acc[v];
        }
    }
    for (int i = lane; i < 16 * 128; i += 32) {  // move h to decoder slot (cols 192..319)
        int a = i >> 7, u = i & 127;
        abuf[a*320 + 192 + u] = abuf[a*320 + 64 + u];
    }
    { int a = lane >> 1; if (lane < 32) shCur[wv][a][lane & 1] = 0.f; }

    const int S_OFF = N_AG * PRED_L * 2;
    const int R_OFF = N_AG * PRED_L * 4;

    // ============================== DECODER ==============================
    for (int t = 0; t < PRED_L; ++t) {
        // emb + pos_feat (decoder social collapses to ONE row per agent: rel = -cur)
        for (int i = lane; i < 16 * 64; i += 32) {
            int a = i >> 6, e = i & 63;
            float cx = shCur[wv][a][0], cy = shCur[wv][a][1];
            abuf[a*320 + e] = (f16)fmaxf(0.f,  cx*shWdW[e] + cy*shWdW[64 + e] + shWdb[e]);
            pf[a*64 + e]    = (f16)fmaxf(0.f, -cx*shSpW[e] - cy*shSpW[64 + e] + shSpb[e]);
        }
        // social row: relu(sH + pf@sWp + sb) * ((|cur|<=R) && any(mask))
        {
            v16h paf[2];
            #pragma unroll
            for (int kt = 0; kt < 2; ++kt) paf[kt] = load_a_frag(pf + kt*32, 64, lane);
            for (int jt = 0; jt < 8; ++jt) {
                v8f acc = vzero();
                #pragma unroll
                for (int kt = 0; kt < 2; ++kt)
                    acc = wmma16(paf[kt], load_b_frag(shWp + (jt*16)*64 + kt*32, 64, lane), acc);
                const int u = jt*16 + col;
                #pragma unroll
                for (int v = 0; v < 8; ++v) {
                    int a = v + 8*kh;
                    float cx = shCur[wv][a][0], cy = shCur[wv][a][1];
                    bool vv = (cx*cx + cy*cy <= 4.0f) && (shVm[wv][a] != 0u);
                    float s = fmaxf(0.f, acc[v] + sH[a*128 + u] + shSb[u]);
                    abuf[a*320 + 64 + u] = (f16)(vv ? s : 0.f);
                }
            }
        }
        // fused gates: [emb|social|h](K=320) @ [dec_Wih;dec_Whh]^T
        {
            v16h gaf[10];
            #pragma unroll
            for (int kt = 0; kt < 10; ++kt) gaf[kt] = load_a_frag(abuf + kt*32, 320, lane);
            for (int j = 0; j < 8; ++j) {
                v8f ai = vzero(), af = vzero(), ag = vzero(), ao = vzero();
                #pragma unroll
                for (int kt = 0; kt < 10; ++kt) {
                    ai = wmma16(gaf[kt], load_b_frag(gDecW + (  0 + j*16)*320 + kt*32, 320, lane), ai);
                    af = wmma16(gaf[kt], load_b_frag(gDecW + (128 + j*16)*320 + kt*32, 320, lane), af);
                    ag = wmma16(gaf[kt], load_b_frag(gDecW + (256 + j*16)*320 + kt*32, 320, lane), ag);
                    ao = wmma16(gaf[kt], load_b_frag(gDecW + (384 + j*16)*320 + kt*32, 320, lane), ao);
                }
                const int u = j*16 + col;
                const float bi = shDecB[u], bf = shDecB[128+u], bg = shDecB[256+u], bo = shDecB[384+u];
                #pragma unroll
                for (int v = 0; v < 8; ++v) {
                    int mrow = v + 8*kh;
                    float cold = cbuf[mrow*128 + u];
                    float cn = sigm(af[v] + bf) * cold + sigm(ai[v] + bi) * tanh_f(ag[v] + bg);
                    cbuf[mrow*128 + u] = cn;
                    float hn = sigm(ao[v] + bo) * tanh_f(cn);
                    htmp[mrow*128 + u] = (f16)hn;
                    soc[mrow*128 + u]  = hn;          // f32 h for the output head
                }
            }
        }
        for (int i = lane; i < 16 * 128; i += 32)     // publish h for next step
            abuf[(i >> 7) * 320 + 192 + (i & 127)] = htmp[i];
        // output head: raw = h @ out_W + out_b  (16x5, VALU)
        for (int idx = lane; idx < 80; idx += 32) {
            int a = idx / 5, k = idx % 5, aG = agBase + a;
            float accu = shOutb[k];
            for (int u2 = 0; u2 < 128; ++u2) accu += soc[a*128 + u2] * shOutW[u2*5 + k];
            if (k < 2) {
                out[(aG*PRED_L + t)*2 + k] = accu + shOrig[wv][a][k];
                shCur[wv][a][k] = accu;                       // carry relative mu
            } else if (k < 4) {
                out[S_OFF + (aG*PRED_L + t)*2 + (k - 2)] = __expf(fminf(4.f, fmaxf(-4.f, accu)));
            } else {
                out[R_OFF + aG*PRED_L + t] = tanh_f(accu);
            }
        }
    }
}

extern "C" void kernel_launch(void* const* d_in, const int* in_sizes, int n_in,
                              void* d_out, int out_size, void* d_ws, size_t ws_size,
                              hipStream_t stream) {
    const float* obs      = (const float*)d_in[0];
    const float* nb_obs   = (const float*)d_in[1];
    const int*   nb_mask  = (const int*)d_in[2];
    const float* we_enc_W = (const float*)d_in[3];
    const float* we_enc_b = (const float*)d_in[4];
    const float* enc_Wih  = (const float*)d_in[5];
    const float* enc_Whh  = (const float*)d_in[6];
    const float* enc_bih  = (const float*)d_in[7];
    const float* enc_bhh  = (const float*)d_in[8];
    const float* sp_pos_W = (const float*)d_in[9];
    const float* sp_pos_b = (const float*)d_in[10];
    const float* sp_soc_W = (const float*)d_in[11];
    const float* sp_soc_b = (const float*)d_in[12];
    const float* we_dec_W = (const float*)d_in[13];
    const float* we_dec_b = (const float*)d_in[14];
    const float* dec_Wih  = (const float*)d_in[15];
    const float* dec_Whh  = (const float*)d_in[16];
    const float* dec_bih  = (const float*)d_in[17];
    const float* dec_bhh  = (const float*)d_in[18];
    const float* out_W    = (const float*)d_in[19];
    const float* out_b    = (const float*)d_in[20];

    prep_kernel<<<288, 256, 0, stream>>>(enc_Wih, enc_Whh, dec_Wih, dec_Whh, sp_soc_W,
                                         enc_bih, enc_bhh, dec_bih, dec_bhh, (char*)d_ws);

    dim3 grid(N_AG / (WPB * AGW));           // 8192 / 64 = 128 blocks
    social_lstm_kernel<<<grid, WPB * 32, 0, stream>>>(
        obs, nb_obs, nb_mask, we_enc_W, we_enc_b, sp_pos_W, sp_pos_b, sp_soc_b,
        we_dec_W, we_dec_b, out_W, out_b, (const char*)d_ws, (float*)d_out);
}
